// GAT_E_to_R_19971597926539
// MI455X (gfx1250) — matrix operand order repaired and checked by reading, not compiled, not run on hardware
//
#include <hip/hip_runtime.h>
#include <math.h>
#include <float.h>

typedef __attribute__((ext_vector_type(16))) __bf16 v16bf;
typedef __attribute__((ext_vector_type(8)))  __bf16 v8bf;
typedef __attribute__((ext_vector_type(8)))  float  v8f;

#define EHID 256
#define CHID 128

__device__ __forceinline__ float leaky(float x) { return x >= 0.f ? x : 0.01f * x; }

// float atomic max via sign-split monotone int trick (maps to
// global_atomic_max_i32 / global_atomic_min_u32)
__device__ __forceinline__ void atomicMaxF(float* addr, float v) {
    if (v >= 0.f) atomicMax((int*)addr, __float_as_int(v));
    else          atomicMin((unsigned int*)addr, __float_as_uint(v));
}

__device__ __forceinline__ float waveSum(float v) {
#pragma unroll
    for (int off = 16; off > 0; off >>= 1) v += __shfl_xor(v, off, 32);
    return v;
}

// ---------------- utility kernels ----------------

__global__ void k_fill(float* p, float v, long n) {
    long i = (long)blockIdx.x * blockDim.x + threadIdx.x;
    long st = (long)gridDim.x * blockDim.x;
    for (; i < n; i += st) p[i] = v;
}

__global__ void k_f32_to_bf16(const float* __restrict__ src, __bf16* __restrict__ dst, long n) {
    long i = (long)blockIdx.x * blockDim.x + threadIdx.x;
    long st = (long)gridDim.x * blockDim.x;
    for (; i < n; i += st) dst[i] = (__bf16)src[i];
}

// Pack B [K x 128(+128)] row-major f32 into per-lane WMMA-fragment bf16 layout (contiguous k order, same as A):
// dst[(((nt*KT + kk)*32 + lane)*16 + e)] = B[kk*32 + (lane<16?0:16) + e][nt*16 + (lane&15)]
__global__ void k_pack_b(const float* __restrict__ W0, const float* __restrict__ W1,
                         __bf16* __restrict__ dst, int K, int NT) {
    int KT = K >> 5;
    int total = NT * KT * 32 * 16;
    int i = blockIdx.x * blockDim.x + threadIdx.x;
    if (i >= total) return;
    int e    = i & 15;
    int lane = (i >> 4) & 31;
    int kk   = (i >> 9) % KT;
    int nt   = (i >> 9) / KT;
    int k = kk * 32 + ((lane < 16) ? 0 : 16) + e;
    int n = nt * 16 + (lane & 15);
    float v = (n < CHID) ? W0[k * CHID + n] : W1[k * CHID + (n - CHID)];
    dst[i] = (__bf16)v;
}

// ---------------- WMMA GEMM, register-blocked over all column tiles ----------------
// One wave owns a 16-row M stripe and ALL NT 16-wide column tiles: each A fragment
// is loaded once per K-step and reused across NT WMMAs (A HBM traffic = 1x).
// __launch_bounds__(128): 4 waves/block -> ~512-VGPR budget per wave, so the
// NT*8 accumulator VGPRs stay in registers (no scratch spills) and the scheduler
// has headroom to prefetch B fragments ahead of the WMMAs.
// A-frag (16x32 bf16): lane l holds row M=l&15; K {0..15} (l<16) or
// {16..31} (l>=16)  -> two b128 loads.
// B-frag: pre-packed 16 contiguous bf16 per lane -> two b128 loads, immediate offsets.
template <int KT, int NT, int LDO, bool RELU, bool WRITE_BF>
__global__ __launch_bounds__(128)
void k_wmma_gemm(const __bf16* __restrict__ A, const __bf16* __restrict__ Bp,
                 float* __restrict__ out, __bf16* __restrict__ out_bf, int M) {
    int wave = (blockIdx.x * blockDim.x + threadIdx.x) >> 5;  // = M tile index
    int lane = threadIdx.x & 31;
    if (wave * 16 >= M) return;
    int hi = lane >> 4;                      // which K-half this lane owns
    int mrow = wave * 16 + (lane & 15);
    const __bf16* Arow = A + (size_t)mrow * (KT * 32);
    const __bf16* Blane = Bp + (size_t)lane * 16;   // all tile/kstep offsets are immediates

    v8f acc[NT];
#pragma unroll
    for (int nt = 0; nt < NT; ++nt) acc[nt] = (v8f){0.f, 0.f, 0.f, 0.f, 0.f, 0.f, 0.f, 0.f};

#pragma unroll
    for (int kk = 0; kk < KT; ++kk) {
        int kbase = kk * 32 + hi * 16;
        v8bf a0 = *reinterpret_cast<const v8bf*>(Arow + kbase);
        v8bf a1 = *reinterpret_cast<const v8bf*>(Arow + kbase + 8);
        v16bf a;
#pragma unroll
        for (int e = 0; e < 8; ++e) { a[e] = a0[e]; a[e + 8] = a1[e]; }
#pragma unroll
        for (int nt = 0; nt < NT; ++nt) {
            const __bf16* bp = Blane + (size_t)(nt * KT + kk) * 32 * 16;
            v8bf b0 = *reinterpret_cast<const v8bf*>(bp);
            v8bf b1 = *reinterpret_cast<const v8bf*>(bp + 8);
            v16bf b;
#pragma unroll
            for (int e = 0; e < 8; ++e) { b[e] = b0[e]; b[e + 8] = b1[e]; }
            acc[nt] = __builtin_amdgcn_wmma_f32_16x16x32_bf16(false, a, false, b,
                                                              (short)0, acc[nt], false, false);
        }
    }

    // C/D layout: lane<16 -> N=lane, rows M=wave*16+{0..7}; lane>=16 -> N=lane-16, rows +8
    int mbase = wave * 16 + hi * 8;
#pragma unroll
    for (int nt = 0; nt < NT; ++nt) {
        int col = nt * 16 + (lane & 15);
#pragma unroll
        for (int r = 0; r < 8; ++r) {
            float v = acc[nt][r];
            if (RELU) v = fmaxf(v, 0.f);
            out[(size_t)(mbase + r) * LDO + col] = v;
            if (WRITE_BF && col < CHID) out_bf[(size_t)(mbase + r) * CHID + col] = (__bf16)v;
        }
    }
}

// ---------------- per-entity / per-relation scalar dots ----------------

__global__ void k_ent_scalars(const float* __restrict__ xr,
                              const float* ah1, const float* ah2, const float* ah3, const float* ah4,
                              const float* at1, const float* at2, const float* at3,
                              float* h1, float* h2, float* h3, float* h4,
                              float* t1o, float* t2o, float* t3o, float* rn, int NE) {
    int wave = (blockIdx.x * blockDim.x + threadIdx.x) >> 5;
    int lane = threadIdx.x & 31;
    if (wave >= NE) return;
    const float* h = xr + (size_t)wave * EHID;       // x_r_h row
    const float* t = h + CHID;                       // x_r_t row
    float s1 = 0, s2 = 0, s3 = 0, s4 = 0, u1 = 0, u2 = 0, u3 = 0, nn = 0;
#pragma unroll
    for (int j = 0; j < 4; ++j) {
        int c = lane + 32 * j;
        float hv = h[c], tv = t[c];
        s1 += hv * ah1[c]; s2 += hv * ah2[c]; s3 += hv * ah3[c]; s4 += hv * ah4[c];
        u1 += tv * at1[c]; u2 += tv * at2[c]; u3 += tv * at3[c]; nn += tv * tv;
    }
    s1 = waveSum(s1); s2 = waveSum(s2); s3 = waveSum(s3); s4 = waveSum(s4);
    u1 = waveSum(u1); u2 = waveSum(u2); u3 = waveSum(u3); nn = waveSum(nn);
    if (lane == 0) {
        h1[wave] = s1; h2[wave] = s2; h3[wave] = s3; h4[wave] = s4;
        t1o[wave] = u1; t2o[wave] = u2; t3o[wave] = u3; rn[wave] = sqrtf(nn);
    }
}

__global__ void k_rel_scalars(const float* __restrict__ r_emb, const float* ar1, const float* ar2,
                              float* rr1, float* rr2, int NR) {
    int wave = (blockIdx.x * blockDim.x + threadIdx.x) >> 5;
    int lane = threadIdx.x & 31;
    if (wave >= NR) return;
    const float* r = r_emb + (size_t)wave * CHID;
    float s1 = 0, s2 = 0;
#pragma unroll
    for (int j = 0; j < 4; ++j) {
        int c = lane + 32 * j;
        s1 += r[c] * ar1[c]; s2 += r[c] * ar2[c];
    }
    s1 = waveSum(s1); s2 = waveSum(s2);
    if (lane == 0) { rr1[wave] = s1; rr2[wave] = s2; }
}

// ---------------- edge / class attention passes ----------------

__global__ void k_edge1(const int* h_idx, const int* t_idx, const int* rel, const int* tn,
                        const float* h1, const float* t1, const float* rr1,
                        float* ebuf, float* m1, int E) {
    int i = blockIdx.x * blockDim.x + threadIdx.x;
    if (i >= E) return;
    float e1 = 0.5f * (h1[h_idx[i]] + t1[t_idx[i]]) + rr1[rel[i]];
    float l = leaky(e1);
    ebuf[i] = l;
    atomicMaxF(&m1[tn[i]], l);
}

// generic exp/sum pass (reused for level-1, level-2, level-3 softmaxes)
__global__ void k_exp_sum(const int* seg, const float* m, float* buf, float* s, int n) {
    int i = blockIdx.x * blockDim.x + threadIdx.x;
    if (i >= n) return;
    int g = seg[i];
    float p = expf(buf[i] - m[g]);
    buf[i] = p;
    atomicAdd(&s[g], p);
}

__global__ void k_edge3(const int* h_idx, const int* t_idx, const int* rel, const int* tn,
                        const int* cls, const float* s1, const float* rn,
                        const float* h2, const float* h3, const float* t2, const float* t3,
                        const float* rr2, float* ebuf, float* scale, float* m2, int E) {
    int i = blockIdx.x * blockDim.x + threadIdx.x;
    if (i >= E) return;
    int ti = t_idx[i];
    float alpha = ebuf[i] / (s1[tn[i]] + 1e-16f);
    float sc = alpha / fmaxf(alpha * rn[ti], 1e-12f);   // x_t row = sc * x_r_t[ti]
    scale[i] = sc;
    float e_t = sc * t2[ti];
    float e_h = h2[h_idx[i]];
    float e_r = 0.5f * (rr2[rel[i]] + 0.5f * (h3[h_idx[i]] + sc * t3[ti]));
    float l = leaky(e_t + e_r + e_h);
    ebuf[i] = l;
    atomicMaxF(&m2[cls[i]], l);
}

// x_class[cls] += beta * x_t[t_idx] where x_t[j] = scale[j]*x_r_t[t_idx[j]]
__global__ void k_spmm(const int* t_idx, const int* cls, const float* ebuf, const float* s2,
                       const float* scale, const float* __restrict__ xr,
                       float* x_class, int E) {
    int wave = (blockIdx.x * blockDim.x + threadIdx.x) >> 5;
    int lane = threadIdx.x & 31;
    if (wave >= E) return;
    int c = cls[wave];
    float beta = ebuf[wave] / (s2[c] + 1e-16f);
    int ti = t_idx[wave];
    float coef = beta * scale[ti];
    int g = t_idx[ti];                                  // t_idx[t_idx[i]]
    const float* src = xr + (size_t)g * EHID + CHID;    // x_r_t[g]
#pragma unroll
    for (int j = 0; j < 4; ++j) {
        int k = lane + 32 * j;
        atomicAdd(&x_class[(size_t)c * CHID + k], coef * src[k]);
    }
}

__global__ void k_class1(const float* __restrict__ x_class, const float* a_c, const int* hc,
                         const float* h4, float* cbuf, float* m3, int NC) {
    int wave = (blockIdx.x * blockDim.x + threadIdx.x) >> 5;
    int lane = threadIdx.x & 31;
    if (wave >= NC) return;
    const float* row = x_class + (size_t)wave * CHID;
    float s = 0;
#pragma unroll
    for (int j = 0; j < 4; ++j) { int c = lane + 32 * j; s += row[c] * a_c[c]; }
    s = waveSum(s);
    if (lane == 0) {
        float l = leaky(s + h4[hc[wave]]);
        cbuf[wave] = l;
        atomicMaxF(&m3[hc[wave]], l);
    }
}

__global__ void k_class3(const float* __restrict__ x_class, const int* hc, const float* cbuf,
                         const float* s3, float* xeh, int NC) {
    int wave = (blockIdx.x * blockDim.x + threadIdx.x) >> 5;
    int lane = threadIdx.x & 31;
    if (wave >= NC) return;
    int h = hc[wave];
    float g = cbuf[wave] / (s3[h] + 1e-16f);
    const float* row = x_class + (size_t)wave * CHID;
#pragma unroll
    for (int j = 0; j < 4; ++j) {
        int k = lane + 32 * j;
        atomicAdd(&xeh[(size_t)h * CHID + k], g * row[k]);
    }
}

// highway combine; d_out currently holds x_e_h (accumulated in place)
__global__ void k_final(const float* __restrict__ gbuf, const float* hw_b,
                        const float* __restrict__ xr, float* out, long n) {
    long i = (long)blockIdx.x * blockDim.x + threadIdx.x;
    long st = (long)gridDim.x * blockDim.x;
    for (; i < n; i += st) {
        int c = (int)(i & (CHID - 1));
        long row = i >> 7;
        float g = 1.f / (1.f + expf(-(gbuf[i] + hw_b[c])));
        float xrh = xr[row * EHID + c];
        out[i] = g * out[i] + (1.f - g) * xrh;
    }
}

// ---------------- host launcher ----------------

extern "C" void kernel_launch(void* const* d_in, const int* in_sizes, int n_in,
                              void* d_out, int out_size, void* d_ws, size_t ws_size,
                              hipStream_t stream) {
    const float* x_e   = (const float*)d_in[0];
    const int*   eidx  = (const int*)d_in[1];
    const int*   rel   = (const int*)d_in[2];
    const int*   tn    = (const int*)d_in[3];
    const float* r_emb = (const float*)d_in[4];
    const int*   cls   = (const int*)d_in[5];
    const int*   hc    = (const int*)d_in[6];
    const float* ah1 = (const float*)d_in[7],  *ah2 = (const float*)d_in[8];
    const float* ah3 = (const float*)d_in[9],  *ah4 = (const float*)d_in[10];
    const float* at1 = (const float*)d_in[11], *at2 = (const float*)d_in[12];
    const float* at3 = (const float*)d_in[13];
    const float* ar1 = (const float*)d_in[14], *ar2 = (const float*)d_in[15];
    const float* a_c = (const float*)d_in[16];
    const float* w_h = (const float*)d_in[17], *w_t = (const float*)d_in[18];
    const float* hw_w = (const float*)d_in[19], *hw_b = (const float*)d_in[20];

    const int NE = in_sizes[0] / EHID;
    const int E  = in_sizes[1] / 2;
    const int NR = in_sizes[4] / CHID;
    const int NC = in_sizes[6];
    const int* h_idx = eidx;
    const int* t_idx = eidx + E;

    char* ws = (char*)d_ws;
    size_t off = 0;
    auto take = [&](size_t bytes) -> char* {
        char* p = ws + off;
        off += (bytes + 255) & ~(size_t)255;
        return p;
    };
    float*  xr      = (float*)take((size_t)NE * EHID * 4);   // [x_r_h | x_r_t] fused, stride 256
    __bf16* xe_bf   = (__bf16*)take((size_t)NE * EHID * 2);  // reused as gbuf (f32 NE*128) later
    __bf16* xrh_bf  = (__bf16*)take((size_t)NE * CHID * 2);
    __bf16* bp_proj = (__bf16*)take((size_t)16 * 8 * 32 * 16 * 2);
    __bf16* bp_hw   = (__bf16*)take((size_t)8 * 4 * 32 * 16 * 2);
    float* h1 = (float*)take((size_t)NE * 4), *h2 = (float*)take((size_t)NE * 4);
    float* h3 = (float*)take((size_t)NE * 4), *h4 = (float*)take((size_t)NE * 4);
    float* t1s = (float*)take((size_t)NE * 4), *t2s = (float*)take((size_t)NE * 4);
    float* t3s = (float*)take((size_t)NE * 4), *rn = (float*)take((size_t)NE * 4);
    float* rr1 = (float*)take((size_t)NR * 4), *rr2 = (float*)take((size_t)NR * 4);
    float* m1 = (float*)take((size_t)NE * 4), *s1 = (float*)take((size_t)NE * 4);
    float* m2 = (float*)take((size_t)NC * 4), *s2 = (float*)take((size_t)NC * 4);
    float* m3 = (float*)take((size_t)NE * 4), *s3 = (float*)take((size_t)NE * 4);
    float* scale = (float*)take((size_t)E * 4);
    float* ebuf  = (float*)take((size_t)E * 4);
    float* cbuf  = (float*)take((size_t)NC * 4);
    float* x_class = (float*)take((size_t)NC * CHID * 4);
    float* gbuf = (float*)xe_bf;       // alias: xe_bf dead after projection GEMM
    float* xeh  = (float*)d_out;       // accumulate x_e_h in place, combine at the end

    const int TB = 256;
    const int TG = 128;                // GEMM block: 4 waves (matches __launch_bounds__)

    // init
    k_fill<<<512, TB, 0, stream>>>(m1, -FLT_MAX, NE);
    k_fill<<<512, TB, 0, stream>>>(m3, -FLT_MAX, NE);
    k_fill<<<128, TB, 0, stream>>>(m2, -FLT_MAX, NC);
    k_fill<<<512, TB, 0, stream>>>(s1, 0.f, NE);
    k_fill<<<512, TB, 0, stream>>>(s3, 0.f, NE);
    k_fill<<<128, TB, 0, stream>>>(s2, 0.f, NC);
    k_fill<<<2048, TB, 0, stream>>>(x_class, 0.f, (long)NC * CHID);
    k_fill<<<4096, TB, 0, stream>>>(xeh, 0.f, (long)NE * CHID);

    // bf16 conversion + weight packing
    k_f32_to_bf16<<<4096, TB, 0, stream>>>(x_e, xe_bf, (long)NE * EHID);
    k_pack_b<<<(16 * 8 * 32 * 16 + TB - 1) / TB, TB, 0, stream>>>(w_h, w_t, bp_proj, EHID, 16);
    k_pack_b<<<(8 * 4 * 32 * 16 + TB - 1) / TB, TB, 0, stream>>>(hw_w, nullptr, bp_hw, CHID, 8);

    // projections: xr = relu(x_e @ [w_h|w_t]);  also write bf16 copy of x_r_h
    long mtiles = (NE + 15) / 16;   // one wave per 16-row stripe (all 16 col tiles)
    k_wmma_gemm<8, 16, 256, true, true>
        <<<(mtiles + 3) / 4, TG, 0, stream>>>(xe_bf, bp_proj, xr, xrh_bf, NE);

    // per-entity / per-relation attention scalars
    k_ent_scalars<<<(NE + 7) / 8, TB, 0, stream>>>(xr, ah1, ah2, ah3, ah4, at1, at2, at3,
                                                   h1, h2, h3, h4, t1s, t2s, t3s, rn, NE);
    k_rel_scalars<<<(NR + 7) / 8, TB, 0, stream>>>(r_emb, ar1, ar2, rr1, rr2, NR);

    // level 1 softmax over triple_num
    k_edge1<<<(E + TB - 1) / TB, TB, 0, stream>>>(h_idx, t_idx, rel, tn, h1, t1s, rr1, ebuf, m1, E);
    k_exp_sum<<<(E + TB - 1) / TB, TB, 0, stream>>>(tn, m1, ebuf, s1, E);

    // level 2 logits + softmax over class_index
    k_edge3<<<(E + TB - 1) / TB, TB, 0, stream>>>(h_idx, t_idx, rel, tn, cls, s1, rn,
                                                  h2, h3, t2s, t3s, rr2, ebuf, scale, m2, E);
    k_exp_sum<<<(E + TB - 1) / TB, TB, 0, stream>>>(cls, m2, ebuf, s2, E);
    k_spmm<<<(E + 7) / 8, TB, 0, stream>>>(t_idx, cls, ebuf, s2, scale, xr, x_class, E);

    // level 3 softmax over head_class, accumulate x_e_h into d_out
    k_class1<<<(NC + 7) / 8, TB, 0, stream>>>(x_class, a_c, hc, h4, cbuf, m3, NC);
    k_exp_sum<<<(NC + TB - 1) / TB, TB, 0, stream>>>(hc, m3, cbuf, s3, NC);
    k_class3<<<(NC + 7) / 8, TB, 0, stream>>>(x_class, hc, cbuf, s3, xeh, NC);

    // highway gate GEMM: gbuf = x_r_h @ hw_w  (bf16 WMMA)
    k_wmma_gemm<4, 8, 128, false, false>
        <<<(mtiles + 3) / 4, TG, 0, stream>>>(xrh_bf, bp_hw, gbuf, nullptr, NE);

    k_final<<<4096, TB, 0, stream>>>(gbuf, hw_b, xr, (float*)d_out, (long)NE * CHID);
}
